// MPNN_82858509075108
// MI455X (gfx1250) — matrix operand order repaired and checked by reading, not compiled, not run on hardware
//
#include <hip/hip_runtime.h>

typedef __attribute__((ext_vector_type(16))) __bf16 v16bf;
typedef __attribute__((ext_vector_type(8)))  float  v8f;
typedef __attribute__((ext_vector_type(8)))  unsigned short ushort8;
typedef __attribute__((ext_vector_type(4)))  unsigned int uint4v;
typedef __attribute__((ext_vector_type(8)))  unsigned int uint8v;

union Frag { v16bf bf; ushort8 u8[2]; };

__device__ __forceinline__ unsigned short f2bf(float f) {
  unsigned u = __float_as_uint(f);
  u += 0x7FFFu + ((u >> 16) & 1u);           // round-to-nearest-even
  return (unsigned short)(u >> 16);
}
__device__ __forceinline__ float bf2f(unsigned short h) {
  return __uint_as_float(((unsigned)h) << 16);
}

// segment descriptors: h = [seg0 | seg1 | ...] in concat order, each zero-padded to padW
struct SegMap {
  int realW[6];
  int padW[6];
  int nseg;
};
struct ConvSegs {
  const unsigned short* src[6];
  int width[6];   // real width
  int pad[6];     // padded (LDS/K) width, multiple of 32
  int mode[6];    // 0=gather by row[], 1=gather by col[], 2=edge manual, 3=edge TDM (width==pad==64)
  int nseg;
  int KP;         // sum of pad
};

// ---------------- small utility kernels ----------------

__global__ void cvt_f32_to_bf16(const float* __restrict__ in,
                                unsigned short* __restrict__ out, long n) {
  long i = (long)blockIdx.x * blockDim.x + threadIdx.x;
  if (i < n) out[i] = f2bf(in[i]);
}

// Repack W [sum(realW) x 64] fp32 (row-major, segment-concat order) into WMMA-B
// fragment order over the segment-padded K axis.
// K-block kb, N-tile t: lane l element j -> (K = kb*32 + (l>>4)*8 + (j<8 ? j : j+8), N = t*16 + (l&15))
__global__ void pack_w(const float* __restrict__ W, SegMap sm, int Kpad,
                       unsigned short* __restrict__ out) {
  int t = blockIdx.x * blockDim.x + threadIdx.x;
  int total = (Kpad >> 5) * 2048;
  if (t >= total) return;
  int kb   = t >> 11;
  int rem  = t & 2047;
  int tile = rem >> 9;
  int lane = (rem >> 4) & 31;
  int j    = rem & 15;
  int K = kb * 32 + ((lane >> 4) * 8) + (j < 8 ? j : j + 8);
  int N = tile * 16 + (lane & 15);
  int realK = -1, po = 0, ro = 0;
  for (int s = 0; s < sm.nseg; ++s) {
    if (K < po + sm.padW[s]) {
      int kl = K - po;
      if (kl < sm.realW[s]) realK = ro + kl;
      break;
    }
    po += sm.padW[s]; ro += sm.realW[s];
  }
  float v = (realK >= 0) ? W[realK * 64 + N] : 0.0f;
  out[t] = f2bf(v);
}

__global__ void count_edges(const int* __restrict__ row, float* __restrict__ cnt, int E) {
  int i = blockIdx.x * blockDim.x + threadIdx.x;
  if (i < E) atomicAdd(&cnt[row[i]], 1.0f);
}

__global__ void node_finalize(const float* __restrict__ sum, const float* __restrict__ cnt,
                              unsigned short* __restrict__ xbf, int total) {
  int i = blockIdx.x * blockDim.x + threadIdx.x;
  if (i < total) {
    float c = fmaxf(cnt[i >> 6], 1.0f);
    xbf[i] = f2bf(fmaxf(sum[i] / c, 0.0f));       // scatter-mean + ReLU
  }
}

// ---------------- fused edge conv ----------------
// 4 waves/block, 32 edges/wave (two 16-row M-tiles sharing every B fragment).
// Contiguous edge-feature segments are DMA'd into LDS by the Tensor Data Mover;
// random node gathers are per-lane loads (node tables live in L2).
// edge_out = relu(h@W1+b1)@W2+b2 ; scatter pre-ReLU into node_sum ; store relu(edge_out).

__launch_bounds__(128)
__global__ void edge_conv_wmma(
    const int* __restrict__ row, const int* __restrict__ col,
    ConvSegs segs,
    const unsigned short* __restrict__ pW1, const float* __restrict__ b1,
    const unsigned short* __restrict__ pW2, const float* __restrict__ b2,
    unsigned short* __restrict__ out_bf, float* __restrict__ out_f32,
    float* __restrict__ node_sum, int E)
{
  extern __shared__ unsigned short dlds[];       // 4 waves x 32 rows x KP (dyn LDS base = offset 0)
  const int wave = threadIdx.x >> 5;
  const int lane = threadIdx.x & 31;
  const int KP   = segs.KP;
  unsigned short* wbase = dlds + wave * 32 * KP;
  const int tile_base = (blockIdx.x * 4 + wave) * 32;
  if (tile_base >= E) return;                    // wave-uniform: EXEC stays full for WMMA

  // ---- stage h segments into LDS ----
  int segoff = 0;
  for (int s = 0; s < segs.nseg; ++s) {
    const int W = segs.width[s], P = segs.pad[s], md = segs.mode[s];
    unsigned short* dst = wbase + 32 * segoff;
    if (md == 3) {
      // Tensor Data Mover: 1-D copy of 32 consecutive 64-wide bf16 edge rows (4 KB).
      // D# group0: count=1, lds_addr, 57-bit global addr, type=2 (ISA ch.8).
      unsigned long long ga = (unsigned long long)(uintptr_t)segs.src[s]
                              + (unsigned long long)tile_base * 128ull;
      unsigned lds_byte = 2u * (unsigned)(wave * 32 * KP + 32 * segoff);
      unsigned td0 = (unsigned)E * 64u;          // tensor length (elements); OOB reads -> 0
      uint4v g0; uint8v g1;
      g0[0] = 1u;                                // count=1, user mode
      g0[1] = lds_byte;
      g0[2] = (unsigned)ga;
      g0[3] = (unsigned)((ga >> 32) & 0x01FFFFFFull) | (2u << 30);   // addr[56:32] | type=2
      g1[0] = 1u << 16;                          // data_size=2B; no pad/iterate/multicast
      g1[1] = (td0 & 0xFFFFu) << 16;             // tensor_dim0[15:0]
      g1[2] = (td0 >> 16) | (1u << 16);          // tensor_dim0[31:16] | tensor_dim1=1
      g1[3] = 2048u << 16;                       // tile_dim0 = 32*64 elements
      g1[4] = 0u;                                // tile_dim1/2 unused
      g1[5] = td0;                               // tensor_dim0_stride
      g1[6] = 0u;
      g1[7] = 0u;
      asm volatile("tensor_load_to_lds %0, %1" :: "s"(g0), "s"(g1) : "memory");
    } else {
      for (int mm = 0; mm < 32; ++mm) {
        int e = tile_base + mm; if (e >= E) e = E - 1;
        int idx = (md == 0) ? row[e] : (md == 1) ? col[e] : e;
        for (int k = lane; k < P; k += 32) {
          unsigned short v = 0;
          if (k < W) v = segs.src[s][(size_t)idx * W + k];
          dst[mm * P + k] = v;
        }
      }
    }
    segoff += P;
  }
  __builtin_amdgcn_s_wait_tensorcnt(0);          // TDM transfers complete
  asm volatile("s_wait_dscnt 0x0" ::: "memory"); // LDS writes visible wave-wide

  const int mrow = lane & 15;
  const int hi   = lane >> 4;
  const int kbA  = hi * 8;                       // ISA 16-bit A layout: K base per half-wave
  const int nIdx = lane & 15;

  // ---- GEMM1: mid = relu(H @ W1 + b1), 32x64 as 2 M-tiles x 4 N-tiles ----
  v8f acc[2][4] = {};
  segoff = 0;
  for (int s = 0; s < segs.nseg; ++s) {
    const int P = segs.pad[s];
    const unsigned short* segbase = wbase + 32 * segoff;
    for (int kk = 0; kk < P; kk += 32) {
      Frag a0, a1;
      const unsigned short* r0 = segbase + mrow * P + kk + kbA;
      a0.u8[0] = *(const ushort8*)(r0);
      a0.u8[1] = *(const ushort8*)(r0 + 16);
      const unsigned short* r1 = segbase + (16 + mrow) * P + kk + kbA;
      a1.u8[0] = *(const ushort8*)(r1);
      a1.u8[1] = *(const ushort8*)(r1 + 16);
      const unsigned short* pb = pW1 + (size_t)((segoff + kk) >> 5) * 2048 + lane * 16;
#pragma unroll
      for (int t = 0; t < 4; ++t) {
        Frag b;
        const ushort8* pb8 = (const ushort8*)(pb + t * 512);
        b.u8[0] = pb8[0]; b.u8[1] = pb8[1];
        acc[0][t] = __builtin_amdgcn_wmma_f32_16x16x32_bf16(
            false, a0.bf, false, b.bf, (short)0, acc[0][t], false, false);
        acc[1][t] = __builtin_amdgcn_wmma_f32_16x16x32_bf16(
            false, a1.bf, false, b.bf, (short)0, acc[1][t], false, false);
      }
    }
    segoff += P;
  }

  // bias + ReLU; round-trip mid (32x64) through LDS to re-fragment as A
  unsigned short* mid = wbase;                   // H fully consumed; safe to overlay
#pragma unroll
  for (int mt = 0; mt < 2; ++mt) {
#pragma unroll
    for (int t = 0; t < 4; ++t) {
      float bv = b1[t * 16 + nIdx];
#pragma unroll
      for (int i = 0; i < 8; ++i) {
        int M = mt * 16 + i + 8 * hi;            // C/D layout: VGPR i, half-wave hi
        mid[M * 64 + t * 16 + nIdx] = f2bf(fmaxf(acc[mt][t][i] + bv, 0.0f));
      }
    }
  }
  asm volatile("s_wait_dscnt 0x0" ::: "memory");

  // ---- GEMM2: out = mid @ W2 + b2 (K = 64) ----
  v8f acc2[2][4] = {};
#pragma unroll
  for (int kk = 0; kk < 64; kk += 32) {
    Frag a0, a1;
    const unsigned short* r0 = mid + mrow * 64 + kk + kbA;
    a0.u8[0] = *(const ushort8*)(r0);
    a0.u8[1] = *(const ushort8*)(r0 + 16);
    const unsigned short* r1 = mid + (16 + mrow) * 64 + kk + kbA;
    a1.u8[0] = *(const ushort8*)(r1);
    a1.u8[1] = *(const ushort8*)(r1 + 16);
    const unsigned short* pb = pW2 + (size_t)(kk >> 5) * 2048 + lane * 16;
#pragma unroll
    for (int t = 0; t < 4; ++t) {
      Frag b;
      const ushort8* pb8 = (const ushort8*)(pb + t * 512);
      b.u8[0] = pb8[0]; b.u8[1] = pb8[1];
      acc2[0][t] = __builtin_amdgcn_wmma_f32_16x16x32_bf16(
          false, a0.bf, false, b.bf, (short)0, acc2[0][t], false, false);
      acc2[1][t] = __builtin_amdgcn_wmma_f32_16x16x32_bf16(
          false, a1.bf, false, b.bf, (short)0, acc2[1][t], false, false);
    }
  }

  // ---- epilogue: scatter pre-ReLU sums, store ReLU'd edge features ----
#pragma unroll
  for (int mt = 0; mt < 2; ++mt) {
    int rr[8];
#pragma unroll
    for (int i = 0; i < 8; ++i) {
      int e = tile_base + mt * 16 + i + 8 * hi;
      rr[i] = (e < E) ? row[e] : 0;
    }
#pragma unroll
    for (int t = 0; t < 4; ++t) {
      int n = t * 16 + nIdx;
      float bv = b2[n];
#pragma unroll
      for (int i = 0; i < 8; ++i) {
        int e = tile_base + mt * 16 + i + 8 * hi;
        if (e < E) {
          float v = acc2[mt][t][i] + bv;
          atomicAdd(&node_sum[(size_t)rr[i] * 64 + n], v);   // pre-ReLU (matches reference)
          float rl = fmaxf(v, 0.0f);
          if (out_bf)  out_bf[(size_t)e * 64 + n]  = f2bf(rl);
          if (out_f32) out_f32[(size_t)e * 64 + n] = rl;
        }
      }
    }
  }
}

// ---------------- tiny head MLP: relu(x@W1+b1)@W2+b2, 64->64->3 ----------------

__launch_bounds__(128)
__global__ void head_mlp(const unsigned short* __restrict__ x,
                         const float* __restrict__ W1, const float* __restrict__ b1,
                         const float* __restrict__ W2, const float* __restrict__ b2,
                         float* __restrict__ out, int N)
{
  int i = blockIdx.x * blockDim.x + threadIdx.x;
  if (i >= N) return;
  float h[64];
#pragma unroll
  for (int j = 0; j < 64; ++j) h[j] = b1[j];
  for (int k = 0; k < 64; ++k) {
    float xv = bf2f(x[(size_t)i * 64 + k]);
#pragma unroll
    for (int j = 0; j < 64; ++j) h[j] = fmaf(xv, W1[k * 64 + j], h[j]);
  }
  float o0 = b2[0], o1 = b2[1], o2 = b2[2];
#pragma unroll
  for (int j = 0; j < 64; ++j) {
    float hv = fmaxf(h[j], 0.0f);
    o0 = fmaf(hv, W2[j * 3 + 0], o0);
    o1 = fmaf(hv, W2[j * 3 + 1], o1);
    o2 = fmaf(hv, W2[j * 3 + 2], o2);
  }
  out[(size_t)i * 3 + 0] = o0;
  out[(size_t)i * 3 + 1] = o1;
  out[(size_t)i * 3 + 2] = o2;
}

// ---------------- launch ----------------

extern "C" void kernel_launch(void* const* d_in, const int* in_sizes, int n_in,
                              void* d_out, int out_size, void* d_ws, size_t ws_size,
                              hipStream_t stream) {
  const float* node_attr = (const float*)d_in[0];
  const float* edge_attr = (const float*)d_in[1];
  const int*   eidx      = (const int*)d_in[2];
  const int N = in_sizes[0] / 3;
  const int E = in_sizes[1] / 3;
  const int* row = eidx;
  const int* col = eidx + E;

  const float* cW1[4] = {(const float*)d_in[3],  (const float*)d_in[7],
                         (const float*)d_in[11], (const float*)d_in[15]};
  const float* cb1[4] = {(const float*)d_in[4],  (const float*)d_in[8],
                         (const float*)d_in[12], (const float*)d_in[16]};
  const float* cW2[4] = {(const float*)d_in[5],  (const float*)d_in[9],
                         (const float*)d_in[13], (const float*)d_in[17]};
  const float* cb2[4] = {(const float*)d_in[6],  (const float*)d_in[10],
                         (const float*)d_in[14], (const float*)d_in[18]};
  const float* mW1 = (const float*)d_in[19];
  const float* mb1 = (const float*)d_in[20];
  const float* mW2 = (const float*)d_in[21];
  const float* mb2 = (const float*)d_in[22];

  const int KPL[4] = {96, 224, 384, 384};      // sum of per-segment padded widths

  // ---- carve workspace ----
  char* p = (char*)d_ws;
  auto carve = [&](size_t bytes) -> char* {
    char* r = p; p += (bytes + 255) & ~(size_t)255; return r;
  };
  unsigned short* pW1[4]; unsigned short* pW2[4];
  for (int l = 0; l < 4; ++l) pW1[l] = (unsigned short*)carve((size_t)(KPL[l] >> 5) * 2048 * 2);
  for (int l = 0; l < 4; ++l) pW2[l] = (unsigned short*)carve((size_t)2 * 2048 * 2);
  unsigned short* nattr_bf = (unsigned short*)carve((size_t)N * 3 * 2);
  unsigned short* eattr_bf = (unsigned short*)carve((size_t)E * 3 * 2);
  unsigned short* xb[4];
  for (int l = 0; l < 4; ++l) xb[l] = (unsigned short*)carve((size_t)N * 64 * 2);
  unsigned short* eb[3];
  for (int l = 0; l < 3; ++l) eb[l] = (unsigned short*)carve((size_t)E * 64 * 2);
  float* node_sum = (float*)carve((size_t)N * 64 * 4);
  float* cnt      = (float*)carve((size_t)N * 4);
  (void)ws_size; (void)n_in; (void)out_size;

  // ---- segment maps (reference concat order: n1[row], n2[row], n1[col], n2[col], eA, eB) ----
  SegMap wmap[4];
  wmap[0] = { {3, 3, 3, 0, 0, 0},      {32, 32, 32, 0, 0, 0},       3 };
  wmap[1] = { {64, 64, 64, 3, 0, 0},   {64, 64, 64, 32, 0, 0},      4 };
  wmap[2] = { {64, 64, 64, 64, 64, 64},{64, 64, 64, 64, 64, 64},    6 };
  wmap[3] = wmap[2];
  SegMap w2map = { {64, 0, 0, 0, 0, 0}, {64, 0, 0, 0, 0, 0}, 1 };

  // ---- pack weights into WMMA B-fragment order (bf16, segment-padded) ----
  for (int l = 0; l < 4; ++l) {
    int tot1 = (KPL[l] >> 5) * 2048;
    pack_w<<<(tot1 + 255) / 256, 256, 0, stream>>>(cW1[l], wmap[l], KPL[l], pW1[l]);
    pack_w<<<(2 * 2048 + 255) / 256, 256, 0, stream>>>(cW2[l], w2map, 64, pW2[l]);
  }
  // ---- convert attrs to bf16 ----
  {
    long a = (long)N * 3, b = (long)E * 3;
    cvt_f32_to_bf16<<<(int)((a + 255) / 256), 256, 0, stream>>>(node_attr, nattr_bf, a);
    cvt_f32_to_bf16<<<(int)((b + 255) / 256), 256, 0, stream>>>(edge_attr, eattr_bf, b);
  }
  // ---- per-node edge counts (layer-invariant) ----
  hipMemsetAsync(cnt, 0, (size_t)N * 4, stream);
  count_edges<<<(E + 255) / 256, 256, 0, stream>>>(row, cnt, E);

  float* out_x = (float*)d_out;
  float* out_e = (float*)d_out + (size_t)N * 3;
  const int egrid = (E + 127) / 128;           // 4 waves/block x 32 edges/wave

  ConvSegs L[4];
  L[0] = { {nattr_bf, nattr_bf, eattr_bf, nullptr, nullptr, nullptr},
           {3, 3, 3, 0, 0, 0}, {32, 32, 32, 0, 0, 0}, {0, 1, 2, 0, 0, 0}, 3, KPL[0] };
  L[1] = { {xb[0], xb[0], eb[0], eattr_bf, nullptr, nullptr},
           {64, 64, 64, 3, 0, 0}, {64, 64, 64, 32, 0, 0}, {0, 1, 3, 2, 0, 0}, 4, KPL[1] };
  L[2] = { {xb[1], xb[0], xb[1], xb[0], eb[1], eb[0]},
           {64, 64, 64, 64, 64, 64}, {64, 64, 64, 64, 64, 64}, {0, 0, 1, 1, 3, 3}, 6, KPL[2] };
  L[3] = { {xb[2], xb[1], xb[2], xb[1], eb[2], eb[1]},
           {64, 64, 64, 64, 64, 64}, {64, 64, 64, 64, 64, 64}, {0, 0, 1, 1, 3, 3}, 6, KPL[3] };
  unsigned short* outbf[4] = { eb[0], eb[1], eb[2], nullptr };
  float* outf[4] = { nullptr, nullptr, nullptr, out_e };

  for (int l = 0; l < 4; ++l) {
    hipMemsetAsync(node_sum, 0, (size_t)N * 64 * 4, stream);
    size_t shbytes = (size_t)4 * 32 * KPL[l] * 2;     // 4 waves x 32 rows x KP bf16
    edge_conv_wmma<<<egrid, 128, shbytes, stream>>>(
        row, col, L[l], pW1[l], cb1[l], pW2[l], cb2[l],
        outbf[l], outf[l], node_sum, E);
    node_finalize<<<(N * 64 + 255) / 256, 256, 0, stream>>>(node_sum, cnt, xb[l], N * 64);
  }

  head_mlp<<<(N + 127) / 128, 128, 0, stream>>>(xb[3], mW1, mb1, mW2, mb2, out_x, N);
}
